// T5Attention_42382737277470
// MI455X (gfx1250) — matrix operand order repaired and checked by reading, not compiled, not run on hardware
//
#include <hip/hip_runtime.h>
#include <hip/hip_bf16.h>

// ---------------------------------------------------------------------------
// T5 encoder self-attention on MI455X (gfx1250), bf16 WMMA with fp32 accum.
//   B=4, S=1024, D=1024, H=16, DK=64, NUM_BUCKETS=32, MAX_DIST=128
// Round 3: K-loop unrolled by 2 so the double-buffered fragment sets
// alternate roles -- eliminates the v_dual_mov_b32 rotation copies the
// compiler emitted for the explicit "cur = nxt" buffer swap.
// ---------------------------------------------------------------------------

#define B_   4
#define S_   1024
#define D_   1024
#define H_   16
#define DK_  64
#define HD_  (H_ * DK_)      // 1024
#define M_   (B_ * S_)       // 4096

typedef __bf16 bf16_t;
typedef __attribute__((ext_vector_type(16))) __bf16 v16bf;
typedef __attribute__((ext_vector_type(8)))  __bf16 v8bf;
typedef __attribute__((ext_vector_type(8)))  float  v8f;

// round-to-nearest-even f32 -> bf16
__device__ __forceinline__ bf16_t f2bf(float f) {
    union { float f; unsigned u; } v; v.f = f;
    unsigned r = v.u + 0x7FFFu + ((v.u >> 16) & 1u);
    unsigned short h = (unsigned short)(r >> 16);
    bf16_t out;
    __builtin_memcpy(&out, &h, 2);
    return out;
}

__device__ __forceinline__ v8f wmma_bf16(v16bf a, v16bf b, v8f c) {
    // v_wmma_f32_16x16x32_bf16: D = A(16x32) * B(32x16) + C(16x16 f32)
    return __builtin_amdgcn_wmma_f32_16x16x32_bf16(
        false, a, false, b, (short)0, c, false, false);
}

// A-matrix 16x32 bf16 fragment from row-major [16 x ld] tile at p.
// lanes 0-15: M=lane, K chunks {0..7, 16..23}; lanes 16-31: M=lane-16,
// K chunks {8..15, 24..31}.  Two 16B loads per lane.
__device__ __forceinline__ v16bf load_frag_a(const bf16_t* p, int ld, int lane) {
    const int m  = lane & 15;
    const int ko = (lane & 16) ? 8 : 0;
    const bf16_t* base = p + (size_t)m * ld + ko;
    v8bf lo = *(const v8bf*)(base);
    v8bf hi = *(const v8bf*)(base + 16);
    v16bf out;
#pragma unroll
    for (int i = 0; i < 8; ++i) { out[i] = lo[i]; out[i + 8] = hi[i]; }
    return out;
}

// B-matrix 32x16 bf16 fragment.  Source stored as Bt[N][K] row-major
// (i.e. column of the logical K x N matrix is contiguous).
// lanes 0-15: N=lane, K chunks {0..7, 16..23}; lanes 16-31: N=lane-16,
// K chunks {8..15, 24..31}.
__device__ __forceinline__ v16bf load_frag_b(const bf16_t* p, int ld, int lane) {
    const int n  = lane & 15;
    const int ko = (lane & 16) ? 8 : 0;
    const bf16_t* base = p + (size_t)n * ld + ko;
    v8bf lo = *(const v8bf*)(base);
    v8bf hi = *(const v8bf*)(base + 16);
    v16bf out;
#pragma unroll
    for (int i = 0; i < 8; ++i) { out[i] = lo[i]; out[i + 8] = hi[i]; }
    return out;
}

// Software-pipelined 16xKdim x Kdimx64 tile GEMM: acc[4] += A * B.
// A row-major [16 x Kdim]; B stored transposed [64][Kdim].
// Unrolled by 2 fragment-sets (K step 64) so the two in-flight buffers
// alternate roles without register copies: while set X computes, set Y loads.
// Requires Kdim % 64 == 0 and Kdim >= 64.
__device__ __forceinline__ void gemm_16x64_pipelined(
        const bf16_t* __restrict__ Arow, int lda,
        const bf16_t* __restrict__ Bt64, int ldb,
        int Kdim, int lane, v8f acc[4]) {
    v16bf a0 = load_frag_a(Arow, lda, lane);
    v16bf b0[4];
#pragma unroll
    for (int t = 0; t < 4; ++t)
        b0[t] = load_frag_b(Bt64 + (size_t)(t * 16) * ldb, ldb, lane);

    int kk = 0;
    for (; kk + 64 < Kdim; kk += 64) {
        // load set(kk+32) while consuming set(kk)
        v16bf a1 = load_frag_a(Arow + kk + 32, lda, lane);
        v16bf b1[4];
#pragma unroll
        for (int t = 0; t < 4; ++t)
            b1[t] = load_frag_b(Bt64 + (size_t)(t * 16) * ldb + kk + 32, ldb, lane);
#pragma unroll
        for (int t = 0; t < 4; ++t)
            acc[t] = wmma_bf16(a0, b0[t], acc[t]);
        // load set(kk+64) while consuming set(kk+32)
        a0 = load_frag_a(Arow + kk + 64, lda, lane);
#pragma unroll
        for (int t = 0; t < 4; ++t)
            b0[t] = load_frag_b(Bt64 + (size_t)(t * 16) * ldb + kk + 64, ldb, lane);
#pragma unroll
        for (int t = 0; t < 4; ++t)
            acc[t] = wmma_bf16(a1, b1[t], acc[t]);
    }
    // tail: sets (kk) and (kk+32) remain
    {
        v16bf a1 = load_frag_a(Arow + kk + 32, lda, lane);
        v16bf b1[4];
#pragma unroll
        for (int t = 0; t < 4; ++t)
            b1[t] = load_frag_b(Bt64 + (size_t)(t * 16) * ldb + kk + 32, ldb, lane);
#pragma unroll
        for (int t = 0; t < 4; ++t)
            acc[t] = wmma_bf16(a0, b0[t], acc[t]);
#pragma unroll
        for (int t = 0; t < 4; ++t)
            acc[t] = wmma_bf16(a1, b1[t], acc[t]);
    }
}

// ---------------------------------------------------------------------------
// Kernel 1: convert X -> bf16 row-major; W{q,k,v,o} -> bf16 transposed [N][K]
// ---------------------------------------------------------------------------
__global__ void t5_convert_kernel(const float* __restrict__ X,
                                  const float* __restrict__ Wq,
                                  const float* __restrict__ Wk,
                                  const float* __restrict__ Wv,
                                  const float* __restrict__ Wo,
                                  bf16_t* __restrict__ Xb,
                                  bf16_t* __restrict__ Wqt,
                                  bf16_t* __restrict__ Wkt,
                                  bf16_t* __restrict__ Wvt,
                                  bf16_t* __restrict__ Wot) {
    size_t gid = (size_t)blockIdx.x * blockDim.x + threadIdx.x;
    const size_t NX = (size_t)M_ * D_;             // 4,194,304
    if (gid < NX) { Xb[gid] = f2bf(X[gid]); return; }
    size_t t  = gid - NX;                          // 4 x 1,048,576
    int    wr = (int)(t >> 20);
    int   idx = (int)(t & 1048575);
    int     n = idx >> 10;
    int     k = idx & 1023;
    const float* src; bf16_t* dst;
    switch (wr) {
        case 0:  src = Wq; dst = Wqt; break;
        case 1:  src = Wk; dst = Wkt; break;
        case 2:  src = Wv; dst = Wvt; break;
        default: src = Wo; dst = Wot; break;
    }
    dst[(size_t)n * 1024 + k] = f2bf(src[(size_t)k * 1024 + n]);
}

// ---------------------------------------------------------------------------
// Kernel 2: QKV projections.  Block = 128 (4 waves); wave w owns the 16-row
// slice row0 = blockIdx.y*64 + w*16 of a 64x64 output block.
//   Q,K stored [B,H,S,DK] bf16 ; V stored transposed [B,H,DK,S] bf16.
// grid = (HD/64, M/64, 3), block = 128
// ---------------------------------------------------------------------------
__global__ void __launch_bounds__(128)
t5_qkv_gemm_kernel(const bf16_t* __restrict__ Xb,
                   const bf16_t* __restrict__ Wqt,
                   const bf16_t* __restrict__ Wkt,
                   const bf16_t* __restrict__ Wvt,
                   bf16_t* __restrict__ Q,
                   bf16_t* __restrict__ K,
                   bf16_t* __restrict__ Vt) {
    const int wave = threadIdx.x >> 5;
    const int lane = threadIdx.x & 31;
    const int row0 = blockIdx.y * 64 + wave * 16;
    const int col0 = blockIdx.x * 64;
    const int z    = blockIdx.z;
    const bf16_t* Wt = (z == 0) ? Wqt : ((z == 1) ? Wkt : Wvt);

    v8f acc[4] = {};
    gemm_16x64_pipelined(Xb + (size_t)row0 * D_, D_,
                         Wt + (size_t)col0 * D_, D_,
                         D_, lane, acc);

    const int hi = lane >> 4;
    const int nn = lane & 15;
#pragma unroll
    for (int t = 0; t < 4; ++t) {
#pragma unroll
        for (int r = 0; r < 8; ++r) {
            int gM = row0 + r + 8 * hi;
            int gN = col0 + t * 16 + nn;
            int b = gM >> 10, s = gM & 1023;       // S = 1024
            int h = gN >> 6,  d = gN & 63;         // DK = 64
            bf16_t val = f2bf(acc[t][r]);
            if (z == 2)
                Vt[(((size_t)(b * H_ + h)) * DK_ + d) * S_ + s] = val;
            else {
                bf16_t* dst = (z == 0) ? Q : K;
                dst[(((size_t)(b * H_ + h)) * S_ + s) * DK_ + d] = val;
            }
        }
    }
}

// ---------------------------------------------------------------------------
// Kernel 3: flash-style attention with T5 relative position bias.
// grid = (S/64, B*H), block = 128 (4 waves; wave w owns 16 query rows)
// ---------------------------------------------------------------------------
__global__ void __launch_bounds__(128)
t5_attn_kernel(const bf16_t* __restrict__ Q,
               const bf16_t* __restrict__ K,
               const bf16_t* __restrict__ Vt,
               const float*  __restrict__ tab,   // [NUM_BUCKETS, H]
               bf16_t* __restrict__ O) {         // [B, S, H*DK]
    __shared__ float sBias[2048];                           // bias(d = k-q)
    __shared__ __align__(16) bf16_t sP[4][16 * 72];         // padded P tiles

    const int bh = blockIdx.y;
    const int b  = bh >> 4;
    const int h  = bh & 15;

    // Per-block bias line: bias[d + 1023] for d = k - q in [-1023, 1023].
    // T5 bidirectional bucketing: nb=16, max_exact=8, log factor 8/ln(16).
    for (int i = threadIdx.x; i < 2047; i += 128) {
        int d   = i - 1023;
        int bkt = (d > 0) ? 16 : 0;
        int rp  = (d < 0) ? -d : d;
        int idx;
        if (rp < 8) idx = rp;
        else {
            int l = 8 + (int)(__logf((float)rp * 0.125f) * 2.8853900817779268f);
            idx = (l < 15) ? l : 15;
        }
        sBias[i] = tab[(bkt + idx) * H_ + h];
    }
    __syncthreads();

    const int wave = threadIdx.x >> 5;
    const int lane = threadIdx.x & 31;
    const int hi   = lane >> 4;
    const int nn   = lane & 15;
    const int q0   = blockIdx.x * 64 + wave * 16;

    const bf16_t* Qb = Q  + (size_t)bh * S_ * DK_;
    const bf16_t* Kb = K  + (size_t)bh * S_ * DK_;
    const bf16_t* Vb = Vt + (size_t)bh * DK_ * S_;

    v16bf qa0 = load_frag_a(Qb + (size_t)q0 * DK_ + 0,  DK_, lane);
    v16bf qa1 = load_frag_a(Qb + (size_t)q0 * DK_ + 32, DK_, lane);

    float rowm[8], rowl[8];
    v8f   oacc[4] = {};
#pragma unroll
    for (int r = 0; r < 8; ++r) { rowm[r] = -1e30f; rowl[r] = 0.0f; }

    bf16_t* pw = &sP[wave][0];

    for (int kb = 0; kb < S_; kb += 64) {
        // ---- issue K fragment loads first (needed soonest) ----
        v16bf kf[4][2];
#pragma unroll
        for (int t = 0; t < 4; ++t) {
            kf[t][0] = load_frag_b(Kb + (size_t)(kb + t * 16) * DK_ + 0,  DK_, lane);
            kf[t][1] = load_frag_b(Kb + (size_t)(kb + t * 16) * DK_ + 32, DK_, lane);
        }
        // ---- issue V fragment loads behind them (in-flight prefetch that
        //      completes during softmax + LDS round-trip) ----
        v16bf vf[4][2];
#pragma unroll
        for (int t = 0; t < 4; ++t) {
            vf[t][0] = load_frag_b(Vb + (size_t)(t * 16) * S_ + kb + 0,  S_, lane);
            vf[t][1] = load_frag_b(Vb + (size_t)(t * 16) * S_ + kb + 32, S_, lane);
        }

        // ---- scores: S = Q * K^T  (16 x 64 tile, 8 WMMAs) ----
        v8f sc[4];
#pragma unroll
        for (int t = 0; t < 4; ++t) {
            v8f z = {};
            z = wmma_bf16(qa0, kf[t][0], z);
            z = wmma_bf16(qa1, kf[t][1], z);
            sc[t] = z;
        }

        // ---- bias + online softmax (per-row, half-wave reductions) ----
#pragma unroll
        for (int r = 0; r < 8; ++r) {
            int qrow = q0 + r + 8 * hi;
            float pm = -1e30f;
#pragma unroll
            for (int t = 0; t < 4; ++t) {
                int kcol = kb + t * 16 + nn;
                float v = sc[t][r] + sBias[kcol - qrow + 1023];
                sc[t][r] = v;
                pm = fmaxf(pm, v);
            }
#pragma unroll
            for (int mk = 8; mk >= 1; mk >>= 1)
                pm = fmaxf(pm, __shfl_xor(pm, mk, 16));
            float nm    = fmaxf(rowm[r], pm);
            float alpha = __expf(rowm[r] - nm);
            rowm[r] = nm;
            float rs = 0.0f;
#pragma unroll
            for (int t = 0; t < 4; ++t) {
                float p = __expf(sc[t][r] - nm);
                sc[t][r] = p;
                rs += p;
            }
#pragma unroll
            for (int mk = 8; mk >= 1; mk >>= 1)
                rs += __shfl_xor(rs, mk, 16);
            rowl[r] = rowl[r] * alpha + rs;
#pragma unroll
            for (int t = 0; t < 4; ++t)
                oacc[t][r] *= alpha;
        }

        // ---- C-layout -> A-layout via padded LDS (stride 72 bf16) ----
#pragma unroll
        for (int t = 0; t < 4; ++t)
#pragma unroll
            for (int r = 0; r < 8; ++r)
                pw[(r + 8 * hi) * 72 + t * 16 + nn] = f2bf(sc[t][r]);
        asm volatile("s_wait_dscnt 0" ::: "memory");   // store->load ordering

        v16bf pa0 = load_frag_a(pw + 0,  72, lane);
        v16bf pa1 = load_frag_a(pw + 32, 72, lane);

        // ---- O += P * V  (V fragments already resident) ----
#pragma unroll
        for (int t = 0; t < 4; ++t) {
            oacc[t] = wmma_bf16(pa0, vf[t][0], oacc[t]);
            oacc[t] = wmma_bf16(pa1, vf[t][1], oacc[t]);
        }
    }

    // ---- normalize and store O as [B, S, H*DK] bf16 ----
#pragma unroll
    for (int r = 0; r < 8; ++r) {
        int qrow = q0 + r + 8 * hi;
        float inv = 1.0f / rowl[r];
#pragma unroll
        for (int t = 0; t < 4; ++t)
            O[((size_t)(b * S_ + qrow)) * HD_ + h * DK_ + t * 16 + nn] =
                f2bf(oacc[t][r] * inv);
    }
}

// ---------------------------------------------------------------------------
// Kernel 4: output projection  out = O(bf16) * Wo  -> fp32 [B,S,D]
// grid = (D/64, M/64), block = 128
// ---------------------------------------------------------------------------
__global__ void __launch_bounds__(128)
t5_out_gemm_kernel(const bf16_t* __restrict__ O,
                   const bf16_t* __restrict__ Wot,
                   float* __restrict__ out) {
    const int wave = threadIdx.x >> 5;
    const int lane = threadIdx.x & 31;
    const int row0 = blockIdx.y * 64 + wave * 16;
    const int col0 = blockIdx.x * 64;

    v8f acc[4] = {};
    gemm_16x64_pipelined(O + (size_t)row0 * HD_, HD_,
                         Wot + (size_t)col0 * HD_, HD_,
                         HD_, lane, acc);

    const int hi = lane >> 4;
    const int nn = lane & 15;
#pragma unroll
    for (int t = 0; t < 4; ++t)
#pragma unroll
        for (int r = 0; r < 8; ++r)
            out[(size_t)(row0 + r + 8 * hi) * D_ + col0 + t * 16 + nn] = acc[t][r];
}

// ---------------------------------------------------------------------------
// Host-side launcher
// ---------------------------------------------------------------------------
extern "C" void kernel_launch(void* const* d_in, const int* in_sizes, int n_in,
                              void* d_out, int out_size, void* d_ws, size_t ws_size,
                              hipStream_t stream) {
    const float* X   = (const float*)d_in[0];   // [B,S,D]
    const float* Wq  = (const float*)d_in[1];   // [D, H*DK]
    const float* Wk  = (const float*)d_in[2];
    const float* Wv  = (const float*)d_in[3];
    const float* Wo  = (const float*)d_in[4];   // [H*DK, D]
    const float* tab = (const float*)d_in[5];   // [NUM_BUCKETS, H]
    float* out = (float*)d_out;                 // [B,S,D] fp32

    // Workspace partition (all offsets 256B-aligned). Total ~50.3 MB.
    char* ws = (char*)d_ws;
    size_t off = 0;
    bf16_t* Xb  = (bf16_t*)(ws + off); off += (size_t)M_ * D_ * 2;   // 8 MiB
    bf16_t* Wqt = (bf16_t*)(ws + off); off += (size_t)D_ * HD_ * 2;  // 2 MiB
    bf16_t* Wkt = (bf16_t*)(ws + off); off += (size_t)D_ * HD_ * 2;
    bf16_t* Wvt = (bf16_t*)(ws + off); off += (size_t)D_ * HD_ * 2;
    bf16_t* Wot = (bf16_t*)(ws + off); off += (size_t)HD_ * D_ * 2;
    bf16_t* Qb  = (bf16_t*)(ws + off); off += (size_t)B_ * H_ * S_ * DK_ * 2; // 8 MiB
    bf16_t* Kb  = (bf16_t*)(ws + off); off += (size_t)B_ * H_ * S_ * DK_ * 2;
    bf16_t* Vtb = (bf16_t*)(ws + off); off += (size_t)B_ * H_ * DK_ * S_ * 2;
    bf16_t* Ob  = (bf16_t*)(ws + off); off += (size_t)M_ * HD_ * 2;           // 8 MiB

    // 1) convert inputs to bf16 (weights transposed to [N][K])
    {
        size_t total = (size_t)M_ * D_ + 4u * 1024u * 1024u;  // 8,388,608
        int blocks = (int)((total + 255) / 256);
        t5_convert_kernel<<<blocks, 256, 0, stream>>>(
            X, Wq, Wk, Wv, Wo, Xb, Wqt, Wkt, Wvt, Wot);
    }

    // 2) Q/K/V projections
    {
        dim3 grid(HD_ / 64, M_ / 64, 3);
        t5_qkv_gemm_kernel<<<grid, 128, 0, stream>>>(Xb, Wqt, Wkt, Wvt, Qb, Kb, Vtb);
    }

    // 3) attention with relative-position bias
    {
        dim3 grid(S_ / 64, B_ * H_);
        t5_attn_kernel<<<grid, 128, 0, stream>>>(Qb, Kb, Vtb, tab, Ob);
    }

    // 4) output projection -> fp32
    {
        dim3 grid(D_ / 64, M_ / 64);
        t5_out_gemm_kernel<<<grid, 128, 0, stream>>>(Ob, Wot, out);
    }

    (void)in_sizes; (void)n_in; (void)out_size; (void)ws_size;
}
